// DenseNeighConsensus_23545010717394
// MI455X (gfx1250) — matrix operand order repaired and checked by reading, not compiled, not run on hardware
//
#include <hip/hip_runtime.h>
#include <stdint.h>

// ---------------------------------------------------------------------------
// CDNA5 (gfx1250, wave32) implicit-GEMM 4D conv via V_WMMA_F32_16X16X32_BF16
// ---------------------------------------------------------------------------

typedef __attribute__((ext_vector_type(16))) __bf16   v16bf;
typedef __attribute__((ext_vector_type(8)))  float    v8f;
typedef __attribute__((ext_vector_type(4)))  uint32_t v4u;

#define G        32
#define PCH      1336336               // 34^4, per-channel padded volume
#define SH1      39304                 // 34^3
#define SW1      1156                  // 34^2
#define SH2      34
#define XDW      1530                  // halo tile: 3*3*10*34 bf16 = 1530 dwords
#define XPAD     1536                  // padded buffer size (dwords), branch-free tail

union FragBF {                         // bit-cast helper for fragment assembly
    v16bf    v;
    uint32_t u[8];
    __bf16   h[16];
    v4u      q[2];
};

// tap index -> LDS offset within the halo tile [i(3)][j(3)][r(10)][s(34)]
// (constexpr so it folds to an immediate when t is a compile-time constant)
__device__ constexpr int tapOff(int t) {
    const int tt = (t < 81) ? t : 0;       // taps >= 81 hit zero weights
    const int ds = tt % 3, dr = (tt / 3) % 3, dq = (tt / 9) % 3, dp = tt / 27;
    return ((dp * 3 + dq) * 10 + dr) * 34 + ds;
}

// fp32 -> padded bf16 (interior only; borders pre-zeroed by memset)
__global__ __launch_bounds__(256) void pad_cast_kernel(const float* __restrict__ x,
                                                       __bf16* __restrict__ xp) {
    size_t i = (size_t)blockIdx.x * 256 + threadIdx.x;
    if (i >= (size_t)4 * G * G * G * G) return;
    int w2 = i & 31, h2 = (i >> 5) & 31, w1 = (i >> 10) & 31, h1 = (i >> 15) & 31;
    int b  = (int)(i >> 20);
    size_t po = (size_t)b * PCH + (size_t)(h1 + 1) * SH1 + (w1 + 1) * SW1
              + (h2 + 1) * SH2 + (w2 + 1);
    xp[po] = (__bf16)x[i];
}

// One conv4d layer (+bias, optional ReLU). PERM=true applies the tap transpose
// (d1,d2,d3,d4)->(d3,d4,d1,d2), realizing the swap-conjugated branch.
// Workgroup = 256 threads = 8 waves; patch = fixed (b,h1,w1), 8 h2 rows, 32 w2.
// Each wave: one h2 row, two 16(M=w2)x16(N=Cout) WMMA tiles, K = CIN*96 (zero-pad).
// Channel halo tiles are double-buffered: next channel's global loads are issued
// before the current channel's WMMAs so memory latency hides under compute.
template <int CIN, int COUT, bool RELU, bool PERM, bool LASTF32, bool ACCUM>
__global__ __launch_bounds__(256) void conv4d_wmma(const __bf16* __restrict__ in,
                                                   const float* __restrict__ Wg,
                                                   const float* __restrict__ bias,
                                                   void* __restrict__ outv) {
    __shared__ __bf16 Wlds[CIN * 16 * 96] __attribute__((aligned(16)));
    __shared__ __bf16 Xlds[2][2 * XPAD] __attribute__((aligned(16)));  // ping-pong

    const int tid  = threadIdx.x;
    const int lane = tid & 31;
    const int wv   = tid >> 5;          // wave id -> h2 row offset (0..7)
    const int half = lane >> 4;         // K-group select (per ISA A/B layouts)
    const int m    = lane & 15;         // M row (A) / N col (B,C,D)

    const int wg  = blockIdx.x;         // 4 * 32 * 32 * 4 = 16384 blocks
    const int h2b = (wg & 3) * 8;
    const int w1  = (wg >> 2) & 31;
    const int h1  = (wg >> 7) & 31;
    const int b   = wg >> 12;

    // ---- stage weights to LDS (fp32 -> bf16, zero pad, optional tap transpose)
    for (int e = tid; e < CIN * 1536; e += 256) {
        const int t   = e % 96;
        const int col = (e / 96) & 15;
        const int ci  = e / 1536;
        __bf16 val = (__bf16)0.f;
        if (t < 81 && col < COUT) {
            int tr = t;
            if (PERM) {
                int ds = t % 3, dr = (t / 3) % 3, dq = (t / 9) % 3, dp = t / 27;
                tr = ((dr * 3 + ds) * 3 + dp) * 3 + dq;
            }
            val = (__bf16)Wg[((size_t)col * CIN + ci) * 81 + tr];
        }
        Wlds[e] = val;
    }

    const v4u* W128 = (const v4u*)Wlds;

    // per-thread staging: 6 dwords cover the 1530-dword tile (tail clamped)
    const size_t tileBase = (size_t)h1 * (SH1 / 2) + (size_t)w1 * (SW1 / 2)
                          + (size_t)h2b * (SH2 / 2);
    auto stage_issue = [&](int ci, uint32_t* regs) {
        const uint32_t* src32 = (const uint32_t*)(in + ((size_t)b * CIN + ci) * PCH);
        #pragma unroll
        for (int k = 0; k < 6; ++k) {
            int e2 = tid + k * 256;
            e2 = (e2 < XDW) ? e2 : (XDW - 1);                // branch-free tail
            const int s2 = e2 % 17, r = (e2 / 17) % 10, ij = e2 / 170;
            const int i = ij / 3, j = ij % 3;
            regs[k] = src32[tileBase + (size_t)i * (SH1 / 2) + j * (SW1 / 2)
                            + r * (SH2 / 2) + s2];
        }
    };
    auto stage_commit = [&](uint32_t* dst, const uint32_t* regs) {
        #pragma unroll
        for (int k = 0; k < 6; ++k) {
            int e2 = tid + k * 256;
            e2 = (e2 < XDW) ? e2 : (XPAD - 6 + k);           // tail -> pad region
            dst[e2] = regs[k];
        }
    };

    v8f acc0 = {0.f, 0.f, 0.f, 0.f, 0.f, 0.f, 0.f, 0.f};
    v8f acc1 = {0.f, 0.f, 0.f, 0.f, 0.f, 0.f, 0.f, 0.f};

    {   // prologue: stage channel 0 into buffer 0
        uint32_t regs[6];
        stage_issue(0, regs);
        stage_commit((uint32_t*)&Xlds[0][0], regs);
    }
    __syncthreads();

    const int ab = wv * 34 + m;                              // per-lane A base

    for (int ci = 0; ci < CIN; ++ci) {
        uint32_t regs[6];
        if (ci + 1 < CIN) stage_issue(ci + 1, regs);         // overlap with WMMAs

        const __bf16* Xb = &Xlds[ci & 1][0];
        #pragma unroll
        for (int ks = 0; ks < 3; ++ks) {
            // ---- B fragment: two aligned 16B LDS vector loads ----
            FragBF B;
            const int bb = (ci * 16 + m) * 12 + ks * 4 + half;  // v4u units
            B.q[0] = W128[bb];
            B.q[1] = W128[bb + 2];

            // ---- A fragments (both w2 tiles); per-half constant tap offsets ----
            FragBF A0, A1;
            if (half == 0) {
                #pragma unroll
                for (int j = 0; j < 8; ++j) {
                    const int kb = (j < 4) ? 0 : 16;
                    #pragma unroll
                    for (int q = 0; q < 2; ++q) {
                        const int o = tapOff(ks * 32 + kb + 2 * (j & 3) + q) + ab;
                        A0.h[2 * j + q] = Xb[o];
                        A1.h[2 * j + q] = Xb[o + 16];
                    }
                }
            } else {
                #pragma unroll
                for (int j = 0; j < 8; ++j) {
                    const int kb = (j < 4) ? 8 : 24;
                    #pragma unroll
                    for (int q = 0; q < 2; ++q) {
                        const int o = tapOff(ks * 32 + kb + 2 * (j & 3) + q) + ab;
                        A0.h[2 * j + q] = Xb[o];
                        A1.h[2 * j + q] = Xb[o + 16];
                    }
                }
            }
            acc0 = __builtin_amdgcn_wmma_f32_16x16x32_bf16(false, A0.v, false, B.v,
                                                           (short)0, acc0, false, false);
            acc1 = __builtin_amdgcn_wmma_f32_16x16x32_bf16(false, A1.v, false, B.v,
                                                           (short)0, acc1, false, false);
        }

        if (ci + 1 < CIN) stage_commit((uint32_t*)&Xlds[(ci + 1) & 1][0], regs);
        __syncthreads();
    }

    // ---- epilogue: C/D layout VGPR r -> M = r (lanes 0-15) / r+8 (lanes 16-31)
    const int h2o = h2b + wv;
    if (LASTF32) {
        float* out = (float*)outv;
        if (m == 0) {
            const float bv = bias[0];
            const size_t base = ((((size_t)b * G + h1) * G + w1) * G + h2o) * G;
            #pragma unroll
            for (int r = 0; r < 8; ++r) {
                const int mr = r + half * 8;
                float v0 = acc0[r] + bv, v1 = acc1[r] + bv;
                if (ACCUM) { out[base + mr] += v0;      out[base + 16 + mr] += v1; }
                else       { out[base + mr]  = v0;      out[base + 16 + mr]  = v1; }
            }
        }
    } else {
        __bf16* out = (__bf16*)outv;
        if (m < COUT) {
            const float bv = bias[m];
            const size_t base = ((size_t)b * COUT + m) * PCH + (size_t)(h1 + 1) * SH1
                              + (w1 + 1) * SW1 + (h2o + 1) * SH2 + 1;
            #pragma unroll
            for (int r = 0; r < 8; ++r) {
                const int mr = r + half * 8;
                float v0 = acc0[r] + bv, v1 = acc1[r] + bv;
                if (RELU) { v0 = fmaxf(v0, 0.f); v1 = fmaxf(v1, 0.f); }
                out[base + mr]      = (__bf16)v0;
                out[base + 16 + mr] = (__bf16)v1;
            }
        }
    }
}

extern "C" void kernel_launch(void* const* d_in, const int* in_sizes, int n_in,
                              void* d_out, int out_size, void* d_ws, size_t ws_size,
                              hipStream_t stream) {
    (void)in_sizes; (void)n_in; (void)out_size;
    const float* x  = (const float*)d_in[0];
    const float* W1 = (const float*)d_in[1];
    const float* b1 = (const float*)d_in[2];
    const float* W2 = (const float*)d_in[3];
    const float* b2 = (const float*)d_in[4];
    const float* W3 = (const float*)d_in[5];
    const float* b3 = (const float*)d_in[6];
    float* out = (float*)d_out;

    // workspace: padded bf16 x (4ch-vol) + two 40ch-vol activation buffers
    char* ws = (char*)d_ws;
    const size_t xp_bytes = (size_t)4 * PCH * 2;
    const size_t off1 = (xp_bytes + 255) & ~(size_t)255;
    const size_t hb_bytes = (size_t)4 * 10 * PCH * 2;
    const size_t off2 = off1 + hb_bytes;
    const size_t total = off2 + hb_bytes;          // ~224.5 MB
    __bf16* xp  = (__bf16*)ws;
    __bf16* h1b = (__bf16*)(ws + off1);
    __bf16* h2b = (__bf16*)(ws + off2);
    if (ws_size < total) return;                   // cannot proceed safely

    hipMemsetAsync(d_ws, 0, total, stream);        // establishes all halo zeros

    const size_t nx = (size_t)4 * G * G * G * G;
    pad_cast_kernel<<<dim3((unsigned)((nx + 255) / 256)), dim3(256), 0, stream>>>(x, xp);

    dim3 grid(16384), block(256);
    // branch A: plain network
    conv4d_wmma<1, 10, true,  false, false, false><<<grid, block, 0, stream>>>(xp,  W1, b1, h1b);
    conv4d_wmma<10, 10, true, false, false, false><<<grid, block, 0, stream>>>(h1b, W2, b2, h2b);
    conv4d_wmma<10, 1, false, false, true,  false><<<grid, block, 0, stream>>>(h2b, W3, b3, out);
    // branch B: swap-conjugated network == tap-transposed weights; accumulate
    conv4d_wmma<1, 10, true,  true,  false, false><<<grid, block, 0, stream>>>(xp,  W1, b1, h1b);
    conv4d_wmma<10, 10, true, true,  false, false><<<grid, block, 0, stream>>>(h1b, W2, b2, h2b);
    conv4d_wmma<10, 1, false, true,  true,  true ><<<grid, block, 0, stream>>>(h2b, W3, b3, out);
}
